// Stgcn_26929444946464
// MI455X (gfx1250) — matrix operand order repaired and checked by reading, not compiled, not run on hardware
//
#include <hip/hip_runtime.h>
#include <hip/hip_bf16.h>

// ---------------- problem constants ----------------
#define N_    8
#define CIN_  3
#define T_    300
#define V_    25
#define P_    3
#define K_    9
#define NCLS_ 60
#define TV_   (T_ * V_)        // 7500
#define NT_   (N_ * T_)        // 2400
#define M_    (N_ * T_ * V_)   // 60000, divisible by 32 and by 25
#define EPS_  1e-5f

typedef __bf16 bf16;
typedef __attribute__((ext_vector_type(16))) __bf16 bf16x16;
typedef __attribute__((ext_vector_type(8)))  __bf16 bf16x8;
typedef __attribute__((ext_vector_type(8)))  float  f32x8;

__device__ __forceinline__ bf16 f2bf(float f) {
  unsigned u = __builtin_bit_cast(unsigned, f);
  u += 0x7FFFu + ((u >> 16) & 1u);                 // round-to-nearest-even
  unsigned short h = (unsigned short)(u >> 16);
  return __builtin_bit_cast(bf16, h);
}

// ---------------- fp32 -> bf16 weight conversion (row-major O x C kept) ----------------
__global__ void cvt_bf16_kernel(const float* __restrict__ src, bf16* __restrict__ dst, int n) {
  int i = blockIdx.x * blockDim.x + threadIdx.x;
  if (i < n) dst[i] = f2bf(src[i]);
}

// ---------------- input BatchNorm1d stats: 75 channels (v*CIN+c), over (N,T) ----------------
__global__ void bnin_stats_kernel(const float* __restrict__ x, float* __restrict__ st) {
  int vc = blockIdx.x;
  int v = vc / CIN_, c = vc % CIN_;
  float s = 0.f, ss = 0.f;
  for (int i = threadIdx.x; i < NT_; i += blockDim.x) {
    int n = i / T_, t = i % T_;
    float val = x[((size_t)(n * CIN_ + c)) * TV_ + t * V_ + v];
    s += val; ss += val * val;
  }
  __shared__ float sh0[256], sh1[256];
  int tid = threadIdx.x;
  sh0[tid] = s; sh1[tid] = ss; __syncthreads();
  for (int w = 128; w > 0; w >>= 1) {
    if (tid < w) { sh0[tid] += sh0[tid + w]; sh1[tid] += sh1[tid + w]; }
    __syncthreads();
  }
  if (tid == 0) {
    float m = sh0[0] / (float)NT_;
    float var = sh1[0] / (float)NT_ - m * m;
    st[2 * vc] = m; st[2 * vc + 1] = rsqrtf(var + EPS_);
  }
}

// ---------------- per-channel BN stats over M (N,T,V), channel-major f32 ----------------
__global__ void stats_kernel(const float* __restrict__ h, float* __restrict__ st) {
  int c = blockIdx.x;
  const float* row = h + (size_t)c * M_;
  float s = 0.f, ss = 0.f;
  for (int i = threadIdx.x; i < M_; i += blockDim.x) { float v = row[i]; s += v; ss += v * v; }
  __shared__ float sh0[256], sh1[256];
  int tid = threadIdx.x;
  sh0[tid] = s; sh1[tid] = ss; __syncthreads();
  for (int w = 128; w > 0; w >>= 1) {
    if (tid < w) { sh0[tid] += sh0[tid + w]; sh1[tid] += sh1[tid + w]; }
    __syncthreads();
  }
  if (tid == 0) {
    float m = sh0[0] / (float)M_;
    float var = sh1[0] / (float)M_ - m * m;
    st[2 * c] = m; st[2 * c + 1] = rsqrtf(var + EPS_);
  }
}

// ---- input BN + fin 1x1 conv (3 -> 64); bf16 out in CHANNEL-LAST (M x 64) layout ----
__global__ void fin_kernel(const float* __restrict__ x, const float* __restrict__ st,
                           const float* __restrict__ g, const float* __restrict__ bb,
                           const float* __restrict__ fw, const float* __restrict__ fb,
                           bf16* __restrict__ out) {
  int m = blockIdx.x * blockDim.x + threadIdx.x;
  if (m >= M_) return;
  int n = m / TV_, r = m % TV_, v = m % V_;
  float xn[CIN_];
#pragma unroll
  for (int c = 0; c < CIN_; ++c) {
    int vc = v * CIN_ + c;
    float xv = x[((size_t)(n * CIN_ + c)) * TV_ + r];
    xn[c] = (xv - st[2 * vc]) * st[2 * vc + 1] * g[vc] + bb[vc];
  }
  bf16* orow = out + (size_t)m * 64;
#pragma unroll
  for (int ob = 0; ob < 8; ++ob) {
    bf16x8 w;
#pragma unroll
    for (int e = 0; e < 8; ++e) {
      int o = ob * 8 + e;
      float acc = fb[o] + fw[o * CIN_] * xn[0] + fw[o * CIN_ + 1] * xn[1]
                + fw[o * CIN_ + 2] * xn[2];
      w[e] = f2bf(acc);
    }
    *(bf16x8*)(orow + ob * 8) = w;
  }
}

// ---------------- WMMA GEMM: D(O x M_) = W(O x C) * X^T, bf16 in / f32 out ----------------
// W row-major (O x C); X channel-last (M_ x C)  =>  A and B fragments are both
// contiguous 32-byte loads per lane.  One wave per 16x16 output tile.
__global__ void gemm_bf16_wmma_kernel(const bf16* __restrict__ Wb, const bf16* __restrict__ Xt,
                                      float* __restrict__ D, int C) {
  const int m0   = blockIdx.x * 16;
  const int o0   = blockIdx.y * 16;
  const int lane = threadIdx.x & 31;
  const int half = lane >> 4;    // selects K sub-range 16*half..16*half+15 (this code's order, same in A and B)
  const int lid  = lane & 15;    // A row / B,D column
  f32x8 acc = {};
  const bf16* wrow = Wb + (size_t)(o0 + lid) * C;           // A: row o0+lid
  const bf16* xrow = Xt + (size_t)(m0 + lid) * C + half * 16; // B: column m0+lid
  for (int k0 = 0; k0 < C; k0 += 32) {
    // A fragment (16x32): element e -> K = k0 + half*16 + e (matches B)
    bf16x8 alo = *(const bf16x8*)(wrow + k0 + half * 16);
    bf16x8 ahi = *(const bf16x8*)(wrow + k0 + half * 16 + 8);
    bf16x16 a = __builtin_shufflevector(alo, ahi, 0, 1, 2, 3, 4, 5, 6, 7,
                                        8, 9, 10, 11, 12, 13, 14, 15);
    // B fragment (32x16): element e -> K = k0 + half*16 + e (contiguous in channel-last)
    bf16x16 b = *(const bf16x16*)(xrow + k0);
    if (k0 + 32 < C)
      __builtin_prefetch(xrow + k0 + 32, 0, 0);             // global_prefetch_b8
    acc = __builtin_amdgcn_wmma_f32_16x16x32_bf16(
        /*neg_a=*/false, a, /*neg_b=*/false, b,
        /*c_mod=*/(short)0, acc, /*reuse_a=*/false, /*reuse_b=*/false);
  }
  // D layout: VGPR j -> row o0 + half*8 + j, column lid (channel-major f32 out)
  float* drow = D + (size_t)(o0 + half * 8) * M_ + m0 + lid;
#pragma unroll
  for (int j = 0; j < 8; ++j) drow[(size_t)j * M_] = acc[j];
}

// ---- spatial graph matmul: zs[c,(n,t),w] (+)= sum_v Y[c,(n,t),v] * (A_p*imp)[v,w] ----
__global__ void graph_kernel(const float* __restrict__ Y, const float* __restrict__ A,
                             const float* __restrict__ imp, float* __restrict__ zs,
                             int Cout, int p, int accumulate) {
  __shared__ float Ae[V_ * V_];
  for (int i = threadIdx.x; i < V_ * V_; i += blockDim.x)
    Ae[i] = A[p * V_ * V_ + i] * imp[i];
  __syncthreads();
  int total = Cout * M_;
  int i = blockIdx.x * blockDim.x + threadIdx.x;
  if (i >= total) return;
  int m = i % M_;
  int w = m % V_;
  const float* yg = Y + (size_t)(i - w);
  float s = accumulate ? zs[i] : 0.f;
#pragma unroll
  for (int v = 0; v < V_; ++v) s += yg[v] * Ae[v * V_ + w];
  zs[i] = s;
}

// ---------------- temporal banded matmul == causal 9-tap sliding sum over t ----------------
__global__ void temporal_kernel(const float* __restrict__ zs, float* __restrict__ h, int Cout) {
  int total = Cout * M_;
  int i = blockIdx.x * blockDim.x + threadIdx.x;
  if (i >= total) return;
  int m = i % M_;
  int t = (m % TV_) / V_;
  int taps = (t + 1 < K_) ? t + 1 : K_;
  float s = 0.f;
  for (int d = 0; d < taps; ++d) s += zs[i - d * V_];
  h[i] = s;
}

// ---------------- fused BN(+ReLU)(+residual)(+ReLU); f32 channel-major out ----------------
__global__ void apply_kernel(const float* __restrict__ h, const float* __restrict__ st,
                             const float* __restrict__ g, const float* __restrict__ b,
                             const float* __restrict__ res, float* __restrict__ outf,
                             int relu_bn, int final_relu) {
  int m = blockIdx.x * blockDim.x + threadIdx.x;
  int c = blockIdx.y;
  if (m >= M_) return;
  size_t i = (size_t)c * M_ + m;
  float v = (h[i] - st[2 * c]) * st[2 * c + 1] * g[c] + b[c];
  if (relu_bn) v = fmaxf(v, 0.f);
  if (res) { v += res[i]; if (final_relu) v = fmaxf(v, 0.f); }
  outf[i] = v;
}

// ---- LDS 32x32 transpose-pack: f32 channel-major (C x M) -> bf16 channel-last (M x C) ----
__global__ void pack_bf16_t_kernel(const float* __restrict__ in, bf16* __restrict__ out, int C) {
  __shared__ bf16 tile[32][33];     // +1 pad to dodge bank conflicts
  int m0 = blockIdx.x * 32, c0 = blockIdx.y * 32;
  int tx = threadIdx.x, ty = threadIdx.y;
  // coalesced reads along m
  for (int cc = ty; cc < 32; cc += 8)
    tile[cc][tx] = f2bf(in[(size_t)(c0 + cc) * M_ + m0 + tx]);
  __syncthreads();
  // coalesced writes along c
  for (int mm = ty; mm < 32; mm += 8)
    out[(size_t)(m0 + mm) * C + c0 + tx] = tile[tx][mm];
}

// ---------------- joint mean-pool over V ----------------
__global__ void pool_kernel(const float* __restrict__ h, float* __restrict__ hm, int Cout) {
  int i = blockIdx.x * blockDim.x + threadIdx.x;
  if (i >= Cout * NT_) return;
  int c = i / NT_, nt = i % NT_;
  const float* p = h + (size_t)c * M_ + (size_t)nt * V_;
  float s = 0.f;
#pragma unroll
  for (int v = 0; v < V_; ++v) s += p[v];
  hm[i] = s * (1.f / (float)V_);
}

// ---------------- classifier 1x1 conv: out[n,o,t] ----------------
__global__ void fout_kernel(const float* __restrict__ hm, const float* __restrict__ fw,
                            const float* __restrict__ fb, float* __restrict__ out, int Cout) {
  int i = blockIdx.x * blockDim.x + threadIdx.x;
  if (i >= N_ * NCLS_ * T_) return;
  int n = i / (NCLS_ * T_);
  int o = (i / T_) % NCLS_;
  int t = i % T_;
  float s = fb[o];
  for (int c = 0; c < Cout; ++c) s += fw[o * Cout + c] * hm[(size_t)c * NT_ + n * T_ + t];
  out[i] = s;
}

// ==================== host orchestration ====================
extern "C" void kernel_launch(void* const* d_in, const int* in_sizes, int n_in,
                              void* d_out, int out_size, void* d_ws, size_t ws_size,
                              hipStream_t stream) {
  (void)in_sizes; (void)n_in; (void)out_size; (void)ws_size;
  const float* x    = (const float*)d_in[0];
  const float* A    = (const float*)d_in[1];
  const float* bng  = (const float*)d_in[2];
  const float* bnb  = (const float*)d_in[3];
  const float* finW = (const float*)d_in[4];
  const float* finb = (const float*)d_in[5];
  const float* imp[4] = {(const float*)d_in[6], (const float*)d_in[7],
                         (const float*)d_in[8], (const float*)d_in[9]};
  const float* W0 = (const float*)d_in[10]; const float* g0 = (const float*)d_in[11]; const float* b0 = (const float*)d_in[12];
  const float* W1 = (const float*)d_in[13]; const float* g1 = (const float*)d_in[14]; const float* b1 = (const float*)d_in[15];
  const float* Wr1= (const float*)d_in[16]; const float* gr1= (const float*)d_in[17]; const float* br1= (const float*)d_in[18];
  const float* W2 = (const float*)d_in[19]; const float* g2 = (const float*)d_in[20]; const float* b2 = (const float*)d_in[21];
  const float* Wr2= (const float*)d_in[22]; const float* gr2= (const float*)d_in[23]; const float* br2= (const float*)d_in[24];
  const float* W3 = (const float*)d_in[25]; const float* g3 = (const float*)d_in[26]; const float* b3 = (const float*)d_in[27];
  const float* foutW = (const float*)d_in[28]; const float* foutb = (const float*)d_in[29];
  float* out = (float*)d_out;

  // ---- workspace carve-out ----
  char* base = (char*)d_ws;
  size_t off = 0;
  auto alloc = [&](size_t bytes) -> char* {
    char* p = base + off;
    off = (off + bytes + 255) & ~(size_t)255;
    return p;
  };
  float* stats = (float*)alloc(4096 * sizeof(float));
  float* stIN  = stats;          // 150 floats
  float* stM   = stats + 256;    // main-layer BN (<=256 ch)
  float* stR   = stats + 1024;   // residual BN
  float* poolb = (float*)alloc((size_t)256 * NT_ * sizeof(float));
  bf16*  wbf   = (bf16*)alloc((size_t)372736 * sizeof(bf16));
  float* actf0 = (float*)alloc((size_t)256 * M_ * sizeof(float));
  float* actf1 = (float*)alloc((size_t)256 * M_ * sizeof(float));
  bf16*  actb0 = (bf16*)alloc((size_t)256 * M_ * sizeof(bf16));  // channel-last M x C
  bf16*  actb1 = (bf16*)alloc((size_t)256 * M_ * sizeof(bf16));  // channel-last M x C
  float* Yb    = (float*)alloc((size_t)256 * M_ * sizeof(float)); // per-p GEMM out, reused as H
  float* zsb   = (float*)alloc((size_t)256 * M_ * sizeof(float));
  float* Rb    = (float*)alloc((size_t)256 * M_ * sizeof(float));

  bf16* W0b  = wbf;               // 192*64
  bf16* W1b  = wbf + 12288;       // 384*64
  bf16* Wr1b = wbf + 36864;       // 128*64
  bf16* W2b  = wbf + 45056;       // 768*128
  bf16* Wr2b = wbf + 143360;      // 256*128
  bf16* W3b  = wbf + 176128;      // 768*256

  auto cvt = [&](const float* s, bf16* d, int n) {
    cvt_bf16_kernel<<<(n + 255) / 256, 256, 0, stream>>>(s, d, n);
  };
  cvt(W0, W0b, 12288);  cvt(W1, W1b, 24576);  cvt(Wr1, Wr1b, 8192);
  cvt(W2, W2b, 98304);  cvt(Wr2, Wr2b, 32768); cvt(W3, W3b, 196608);

  // input BN stats + fused fin conv -> bf16 channel-last activations (M x 64)
  bnin_stats_kernel<<<V_ * CIN_, 256, 0, stream>>>(x, stIN);
  fin_kernel<<<(M_ + 255) / 256, 256, 0, stream>>>(x, stIN, bng, bnb, finW, finb, actb0);

  auto gemm = [&](const bf16* Wp, const bf16* X, float* D, int C, int O) {
    dim3 g(M_ / 16, O / 16);
    gemm_bf16_wmma_kernel<<<g, 32, 0, stream>>>(Wp, X, D, C);
  };
  auto pack = [&](const float* srcF, bf16* dstB, int C) {
    dim3 g(M_ / 32, C / 32);
    pack_bf16_t_kernel<<<g, dim3(32, 8), 0, stream>>>(srcF, dstB, C);
  };

  auto run_layer = [&](const bf16* Wmain, int Cin, int Cout, const float* impL,
                       const float* gg, const float* bbv, const bf16* inB,
                       const float* res, float* outF, bf16* outB, int final_relu) {
    int tot = Cout * M_;
    for (int p = 0; p < P_; ++p) {
      gemm(Wmain + (size_t)p * Cout * Cin, inB, Yb, Cin, Cout);
      graph_kernel<<<(tot + 255) / 256, 256, 0, stream>>>(Yb, A, impL, zsb, Cout, p, p > 0 ? 1 : 0);
    }
    temporal_kernel<<<(tot + 255) / 256, 256, 0, stream>>>(zsb, Yb, Cout);  // H aliases Yb
    stats_kernel<<<Cout, 256, 0, stream>>>(Yb, stM);
    dim3 ga((M_ + 255) / 256, Cout);
    apply_kernel<<<ga, 256, 0, stream>>>(Yb, stM, gg, bbv, res, outF, 1, final_relu);
    if (outB) pack(outF, outB, Cout);
  };

  // layer 0: 64->64, no residual, no final relu
  run_layer(W0b, 64, 64, imp[0], g0, b0, actb0, nullptr, actf1, actb1, 0);

  // layer 1: 64->128, conv+BN residual
  gemm(Wr1b, actb1, Rb, 64, 128);
  stats_kernel<<<128, 256, 0, stream>>>(Rb, stR);
  { dim3 ga((M_ + 255) / 256, 128);
    apply_kernel<<<ga, 256, 0, stream>>>(Rb, stR, gr1, br1, nullptr, Rb, 0, 0); }
  run_layer(W1b, 64, 128, imp[1], g1, b1, actb1, Rb, actf0, actb0, 1);

  // layer 2: 128->256, conv+BN residual
  gemm(Wr2b, actb0, Rb, 128, 256);
  stats_kernel<<<256, 256, 0, stream>>>(Rb, stR);
  { dim3 ga((M_ + 255) / 256, 256);
    apply_kernel<<<ga, 256, 0, stream>>>(Rb, stR, gr2, br2, nullptr, Rb, 0, 0); }
  run_layer(W2b, 128, 256, imp[2], g2, b2, actb0, Rb, actf1, actb1, 1);

  // layer 3: 256->256, identity residual (layer-2 f32 output); no bf16 consumer
  run_layer(W3b, 256, 256, imp[3], g3, b3, actb1, actf1, actf0, nullptr, 1);

  // mean-pool over joints + classifier
  { int tot = 256 * NT_;
    pool_kernel<<<(tot + 255) / 256, 256, 0, stream>>>(actf0, poolb, 256); }
  { int tot = N_ * NCLS_ * T_;
    fout_kernel<<<(tot + 255) / 256, 256, 0, stream>>>(poolb, foutW, foutb, out, 256); }
}
